// lossEvaluator_38998303047728
// MI455X (gfx1250) — matrix operand order repaired and checked by reading, not compiled, not run on hardware
//
#include <hip/hip_runtime.h>

// ---------------------------------------------------------------------------
// lossEvaluator: simMM = imFtr[B,P,D] x disFtr[B,D]^T -> max over P -> hinge
// B=256, P=64, D=1024.  GEMM (16384x256x1024) via V_WMMA_F32_16X16X32_BF16.
//
// Pass 0: disFtr f32 -> bf16 once (0.5 MB, L2-resident).
// Pass 1: per-image WG (256 thr = 8 wave32).  B tiles staged into LDS with
//         GLOBAL_LOAD_ASYNC_TO_LDS_B128 (ASYNCcnt-tracked, double-buffered,
//         pre-swizzled into WMMA fragment order); B fragments are register
//         double-buffered across the tile loop so WMMAs wait only on the
//         previous ds_load pair.  imFtr streamed from HBM exactly once,
//         converted f32->bf16 in-register.
// Pass 2: 256x256 hinge-loss reduction (single block).
// ---------------------------------------------------------------------------

typedef __attribute__((ext_vector_type(16))) __bf16 bf16x16;
typedef __attribute__((ext_vector_type(8)))  __bf16 bf16x8;
typedef __attribute__((ext_vector_type(4)))  __bf16 bf16x4;
typedef __attribute__((ext_vector_type(8)))  float  f32x8;

#ifndef LOSS_MARGIN
#define LOSS_MARGIN 0.1f
#endif

__device__ __forceinline__ bf16x16 pack_bf16(float4 a, float4 b, float4 c, float4 d) {
    bf16x16 r;
    r[0]  = (__bf16)a.x; r[1]  = (__bf16)a.y; r[2]  = (__bf16)a.z; r[3]  = (__bf16)a.w;
    r[4]  = (__bf16)b.x; r[5]  = (__bf16)b.y; r[6]  = (__bf16)b.z; r[7]  = (__bf16)b.w;
    r[8]  = (__bf16)c.x; r[9]  = (__bf16)c.y; r[10] = (__bf16)c.z; r[11] = (__bf16)c.w;
    r[12] = (__bf16)d.x; r[13] = (__bf16)d.y; r[14] = (__bf16)d.z; r[15] = (__bf16)d.w;
    return r;
}

// ---- Pass 0: disFtr f32 -> bf16 (row-major [256][1024]) --------------------
__global__ __launch_bounds__(256, 4) void cvt_dis_kernel(
    const float* __restrict__ dis, __bf16* __restrict__ out)
{
    const int idx = blockIdx.x * blockDim.x + threadIdx.x;  // 65536 thr, 4 el each
    const float4 v = ((const float4*)dis)[idx];
    bf16x4 o;
    o[0] = (__bf16)v.x; o[1] = (__bf16)v.y; o[2] = (__bf16)v.z; o[3] = (__bf16)v.w;
    ((bf16x4*)out)[idx] = o;
}

// ---- Pass 1: GEMM + max over proposals -------------------------------------
// One workgroup per image b.  256 threads = 8 wave32s.
// Wave (rg, cg): rows p in [rg*16, rg*16+16), cols c in [cg*128, cg*128+128)
// as 8 WMMA tiles of 16x16.
//
// LDS B buffer layout (per K-step, per buffer): 16 column-tiles (ct=0..15);
// block32[ct][lane] = 32 bytes = the 16 bf16 this lane feeds to the WMMA B
// operand for tile ct: col = ct*16 + (lane&15), K = k0 + (lane>>4)*16 .. +15.
__global__ __launch_bounds__(256, 1) void simmax_wmma_kernel(
    const float*  __restrict__ imFtr,     // [B, P, D] f32
    const __bf16* __restrict__ disBf16,   // [B, D]    bf16
    float* __restrict__ simMax)           // [B, B]    f32
{
    constexpr int B = 256, P = 64, D = 1024;
    __shared__ __align__(16) __bf16 Bs[2][16 * 512];  // 2 x 16 KB
    __shared__ float smax[4][B];

    const int tid  = threadIdx.x;
    const int wave = tid >> 5;
    const int lane = tid & 31;
    const int rg   = wave >> 1;   // row group 0..3
    const int cg   = wave & 1;    // col group 0..1
    const int l16  = lane & 15;
    const int hi   = lane >> 4;   // lane half selects K sub-range
    const int b    = blockIdx.x;

    // A: lane l16 holds proposal row (rg*16 + l16)
    const float* arow = imFtr + ((size_t)b * P + (size_t)(rg * 16 + l16)) * D;

    // Async-copy issue: 1024 b128 transfers per K-step, 4 per thread.
    // i = tid*4+j: ct = i>>6, L = (i&63)>>1, h = i&1
    //   src = disBf16[ct*16 + (L&15)][k0 + (L>>4)*16 + h*8]  (8 bf16 = 16 B)
    //   dst = Bs[buf] + ct*512 + L*16 + h*8                  (elements)
    const int i0  = tid * 4;
    const int ct0 = i0 >> 6;
    const int L0  = (i0 & 63) >> 1;

    #define ISSUE_B_ASYNC(nbuf, kk)                                               \
        {                                                                         \
            _Pragma("unroll")                                                     \
            for (int j = 0; j < 4; ++j) {                                         \
                const int Lj = L0 + (j >> 1);                                     \
                const int hj = j & 1;                                             \
                const __bf16* src = disBf16 +                                     \
                    (size_t)(ct0 * 16 + (Lj & 15)) * D + (kk) + (Lj >> 4) * 16 + hj * 8; \
                unsigned dst = (unsigned)(size_t)(&Bs[(nbuf)][ct0 * 512 + Lj * 16 + hj * 8]); \
                asm volatile("global_load_async_to_lds_b128 %0, %1, off"          \
                             :: "v"(dst), "v"(src) : "memory");                   \
            }                                                                     \
        }

    f32x8 acc[8] = {};

    // Prologue: stage first B tile into buffer 0.
    ISSUE_B_ASYNC(0, 0);

    for (int k0 = 0; k0 < D; k0 += 32) {
        const int buf = (k0 >> 5) & 1;

        // My async writes into Bs[buf] are done ...
        asm volatile("s_wait_asynccnt 0x0" ::: "memory");
        // ... and everyone else's too; also all prior reads of Bs[buf^1] done.
        __syncthreads();

        // Overlap: stage next K-step while this one computes.
        if (k0 + 32 < D) ISSUE_B_ASYNC(buf ^ 1, k0 + 32);

        // ---- A fragment (16x32 bf16): contiguous K order, same as B ----
        // lanes 0-15 : elems 0..15 = K k0+0..15
        // lanes 16-31: elems 0..15 = K k0+16..31
        const float* ap0 = arow + k0 + hi * 16;
        const float* ap1 = arow + k0 + hi * 16 + 8;
        float4 a0 = *(const float4*)(ap0);
        float4 a1 = *(const float4*)(ap0 + 4);
        float4 a2 = *(const float4*)(ap1);
        float4 a3 = *(const float4*)(ap1 + 4);
        bf16x16 afrag = pack_bf16(a0, a1, a2, a3);

        if (k0 + 32 < D) __builtin_prefetch(arow + k0 + 32, 0, 1);

        // B fragments: register double-buffered over the 8 tiles so each WMMA
        // only waits for the ds_load pair issued one tile earlier.
        const __bf16* bs = &Bs[buf][0] + (size_t)(cg * 8) * 512 + lane * 16;
        bf16x8 blo = ((const bf16x8*)bs)[0];
        bf16x8 bhi = ((const bf16x8*)bs)[1];
        #pragma unroll
        for (int t = 0; t < 8; ++t) {
            bf16x8 nlo = blo, nhi = bhi;
            if (t < 7) {
                const bf16x8* pn = (const bf16x8*)(bs + (size_t)(t + 1) * 512);
                nlo = pn[0];
                nhi = pn[1];
            }
            bf16x16 bfrag = __builtin_shufflevector(blo, bhi,
                0, 1, 2, 3, 4, 5, 6, 7, 8, 9, 10, 11, 12, 13, 14, 15);

            acc[t] = __builtin_amdgcn_wmma_f32_16x16x32_bf16(
                /*neg_a=*/false, afrag, /*neg_b=*/false, bfrag,
                /*c_mod=*/(short)0, acc[t], /*reuse_a=*/false, /*reuse_b=*/false);

            blo = nlo;
            bhi = nhi;
        }
    }
    #undef ISSUE_B_ASYNC

    // Max over proposals.  C/D layout: VGPR v, lane L -> M = v + 8*(L>=16), N = L%16.
    #pragma unroll
    for (int t = 0; t < 8; ++t) {
        float m = acc[t][0];
        #pragma unroll
        for (int v = 1; v < 8; ++v) m = fmaxf(m, acc[t][v]);
        // combine the two lane halves (rows +0..7 vs +8..15)
        m = fmaxf(m, __shfl_xor(m, 16, 32));
        if (hi == 0) smax[rg][cg * 128 + t * 16 + l16] = m;
    }
    __syncthreads();

    // tid == column index (exactly 256 threads)
    {
        float m = fmaxf(fmaxf(smax[0][tid], smax[1][tid]),
                        fmaxf(smax[2][tid], smax[3][tid]));
        simMax[(size_t)b * B + tid] = m;
    }
}

// ---- Pass 2: hinge loss over the 256x256 simMax ----------------------------
__global__ __launch_bounds__(256, 1) void hinge_loss_kernel(
    const float* __restrict__ simMax,  // [B, B]
    const int*   __restrict__ lbl,     // [B]
    float* __restrict__ out)           // [1]
{
    constexpr int B = 256;
    __shared__ float posS[B];
    __shared__ int   lblS[B];
    __shared__ float rA[B], rB[B], rC[B];

    const int i = threadIdx.x;
    posS[i] = simMax[(size_t)i * B + i];
    lblS[i] = lbl[i];
    __syncthreads();

    const float pi = posS[i];
    const int   li = lblS[i];
    float la = 0.f, lb = 0.f, cnt = 0.f;
    for (int j = 0; j < B; ++j) {
        float s = simMax[(size_t)i * B + j];
        if (li != lblS[j]) {
            cnt += 1.f;
            la  += fmaxf(s - pi      + LOSS_MARGIN, 0.f);  // image -> text
            lb  += fmaxf(s - posS[j] + LOSS_MARGIN, 0.f);  // text  -> image
        }
    }
    rA[i] = la; rB[i] = lb; rC[i] = cnt;
    __syncthreads();

    for (int off = B / 2; off > 0; off >>= 1) {
        if (i < off) { rA[i] += rA[i + off]; rB[i] += rB[i + off]; rC[i] += rC[i + off]; }
        __syncthreads();
    }
    if (i == 0) {
        const float pairNum = rC[0];
        float loss = rA[0] / pairNum;
        if (pairNum > 0.f) loss += rB[0] / pairNum;
        out[0] = loss;
    }
}

extern "C" void kernel_launch(void* const* d_in, const int* in_sizes, int n_in,
                              void* d_out, int out_size, void* d_ws, size_t ws_size,
                              hipStream_t stream) {
    (void)in_sizes; (void)n_in; (void)out_size; (void)ws_size;
    constexpr int B = 256, D = 1024;
    const float* imFtr  = (const float*)d_in[0];   // [256, 64, 1024] f32
    const float* disFtr = (const float*)d_in[1];   // [256, 1024]     f32
    const int*   lbl    = (const int*)d_in[2];     // [256]           i32
    float*       out    = (float*)d_out;           // [1]             f32

    // Workspace layout: [0, 256KB) simMax f32; [256KB, 768KB) disFtr bf16.
    float*  simMax   = (float*)d_ws;
    __bf16* disBf16  = (__bf16*)((char*)d_ws + (size_t)B * B * sizeof(float));

    cvt_dis_kernel<<<dim3((B * D / 4) / 256), dim3(256), 0, stream>>>(disFtr, disBf16);
    simmax_wmma_kernel<<<dim3(B), dim3(256), 0, stream>>>(imFtr, disBf16, simMax);
    hinge_loss_kernel<<<dim3(1), dim3(256), 0, stream>>>(simMax, lbl, out);
}
